// LengthRegulator_48180943127141
// MI455X (gfx1250) — compile-verified, matrix operand order checked by reading, not run on hardware
//
#include <hip/hip_runtime.h>

typedef __attribute__((ext_vector_type(16))) _Float16 v16h;
typedef __attribute__((ext_vector_type(8)))  _Float16 v8h;
typedef __attribute__((ext_vector_type(4)))  _Float16 v4h;
typedef __attribute__((ext_vector_type(8)))  float    v8f;

constexpr int B_     = 32;
constexpr int L_     = 1024;
constexpr int C_     = 256;
constexpr int KTOT   = 768;          // 3 * 256 im2col K
constexpr int DURMAX = 6;
constexpr int MAXLEN = L_ * DURMAX;  // 6144
constexpr int MTOT   = B_ * L_;      // 32768

// ---------------------------------------------------------------------------
// fp32 -> f16 elementwise (vectorized x4)
// ---------------------------------------------------------------------------
__global__ void cvt_f16_kernel(const float* __restrict__ in,
                               _Float16* __restrict__ out, int n4) {
  int i = blockIdx.x * blockDim.x + threadIdx.x;
  if (i < n4) {
    float4 f = ((const float4*)in)[i];
    v4h h = { (_Float16)f.x, (_Float16)f.y, (_Float16)f.z, (_Float16)f.w };
    ((v4h*)out)[i] = h;
  }
}

// ---------------------------------------------------------------------------
// weight convert + transpose: w[k][n] fp32 -> wt[n][k] f16   (k<768, n<256)
// ---------------------------------------------------------------------------
__global__ void cvt_wt_kernel(const float* __restrict__ w,
                              _Float16* __restrict__ wt) {
  int i = blockIdx.x * blockDim.x + threadIdx.x;
  if (i < KTOT * C_) {
    int k = i / C_, n = i % C_;
    wt[(size_t)n * KTOT + k] = (_Float16)w[i];
  }
}

// ---------------------------------------------------------------------------
// im2col conv GEMM via v_wmma_f32_16x16x32_f16.
// Ain:  [B_, L_, C_] f16 activations
// Wt:   [C_out=256][KTOT=768] f16 (transposed weights)
// Out:  [MTOT, C_] f32  (= conv + bias, pre-LayerNorm)
// grid: (MTOT/128, 2), block: 256 (8 waves; wave -> 16(M) x 128(N) tile)
// ---------------------------------------------------------------------------
__global__ __launch_bounds__(256) void conv_gemm_wmma(
    const _Float16* __restrict__ Ain, const _Float16* __restrict__ Wt,
    const float* __restrict__ bias, float* __restrict__ Out) {
  const int lane  = threadIdx.x & 31;
  const int wave  = threadIdx.x >> 5;
  const int m0    = blockIdx.x * 128 + wave * 16;
  const int nbase = blockIdx.y * 128;
  const int nl    = lane & 15;   // M row (A) / N col (B,D) within tile
  const int kh    = lane >> 4;   // K half selector per ISA 16-bit layout

  const int bb   = m0 / L_;          // 16 | 1024 -> whole tile in one batch
  const int lrow = (m0 % L_) + nl;   // this lane's phoneme position

  v8f acc[8];
#pragma unroll
  for (int nt = 0; nt < 8; ++nt) acc[nt] = (v8f){0.f,0.f,0.f,0.f,0.f,0.f,0.f,0.f};

  for (int k0 = 0; k0 < KTOT; k0 += 32) {
    // ---- A fragment (im2col): K-block [k0,k0+31] lies in one kernel tap ----
    const int kw   = k0 >> 8;        // tap index 0..2
    const int cin  = k0 & 255;       // input-channel base
    const int srcl = lrow + kw - 1;  // SAME padding
    const bool ok  = (srcl >= 0) && (srcl < L_);
    const _Float16* ap =
        Ain + ((size_t)bb * L_ + (ok ? srcl : 0)) * C_ + cin + 8 * kh;
    v8h a1 = {}, a2 = {};
    if (ok) { a1 = *(const v8h*)ap; a2 = *(const v8h*)(ap + 16); }
    v16h af;
#pragma unroll
    for (int e = 0; e < 8; ++e) { af[e] = a1[e]; af[8 + e] = a2[e]; }

    // ---- 8 N-subtiles: B fragments from transposed weights ----
#pragma unroll
    for (int nt = 0; nt < 8; ++nt) {
      const _Float16* wp =
          Wt + (size_t)(nbase + nt * 16 + nl) * KTOT + k0 + 8 * kh;
      v8h b1 = *(const v8h*)wp;
      v8h b2 = *(const v8h*)(wp + 16);
      v16h bf;
#pragma unroll
      for (int e = 0; e < 8; ++e) { bf[e] = b1[e]; bf[8 + e] = b2[e]; }

      acc[nt] = __builtin_amdgcn_wmma_f32_16x16x32_f16(
          false, af, false, bf, (short)0, acc[nt], false, false);
    }
  }

  // ---- epilogue: D layout lanes0-15 M=r, lanes16-31 M=r+8 ----
#pragma unroll
  for (int nt = 0; nt < 8; ++nt) {
    const int n  = nbase + nt * 16 + nl;
    const float bv = bias[n];
#pragma unroll
    for (int r = 0; r < 8; ++r) {
      const int m = m0 + r + 8 * kh;
      Out[(size_t)m * C_ + n] = acc[nt][r] + bv;
    }
  }
}

// ---------------------------------------------------------------------------
// LayerNorm + ReLU -> f16, one wave per 256-wide row
// ---------------------------------------------------------------------------
__global__ __launch_bounds__(256) void ln_relu_f16_kernel(
    const float* __restrict__ H, const float* __restrict__ g,
    const float* __restrict__ be, _Float16* __restrict__ Outh) {
  const int lane = threadIdx.x & 31;
  const int row  = blockIdx.x * 8 + (threadIdx.x >> 5);
  const float* p = H + (size_t)row * C_ + lane * 8;
  float4 a = *(const float4*)p, c = *(const float4*)(p + 4);
  float v[8] = {a.x, a.y, a.z, a.w, c.x, c.y, c.z, c.w};
  float s = 0.f, s2 = 0.f;
#pragma unroll
  for (int j = 0; j < 8; ++j) { s += v[j]; s2 += v[j] * v[j]; }
  for (int off = 16; off > 0; off >>= 1) {
    s  += __shfl_xor(s, off, 32);
    s2 += __shfl_xor(s2, off, 32);
  }
  const float mu = s * (1.f / C_);
  const float rs = rsqrtf(s2 * (1.f / C_) - mu * mu + 1e-5f);
  const float4 gg = *(const float4*)(g + lane * 8);
  const float4 g2 = *(const float4*)(g + lane * 8 + 4);
  const float4 bb = *(const float4*)(be + lane * 8);
  const float4 b2 = *(const float4*)(be + lane * 8 + 4);
  const float gv[8] = {gg.x, gg.y, gg.z, gg.w, g2.x, g2.y, g2.z, g2.w};
  const float bv[8] = {bb.x, bb.y, bb.z, bb.w, b2.x, b2.y, b2.z, b2.w};
  v8h o;
#pragma unroll
  for (int j = 0; j < 8; ++j)
    o[j] = (_Float16)fmaxf((v[j] - mu) * rs * gv[j] + bv[j], 0.f);
  *(v8h*)(Outh + (size_t)row * C_ + lane * 8) = o;
}

// ---------------------------------------------------------------------------
// LayerNorm + ReLU + linear(256->1) + ReLU + exp -> durations
// ---------------------------------------------------------------------------
__global__ __launch_bounds__(256) void ln_lin_exp_kernel(
    const float* __restrict__ H, const float* __restrict__ g,
    const float* __restrict__ be, const float* __restrict__ lw,
    const float* __restrict__ lb, float* __restrict__ dur) {
  const int lane = threadIdx.x & 31;
  const int row  = blockIdx.x * 8 + (threadIdx.x >> 5);
  const float* p = H + (size_t)row * C_ + lane * 8;
  float4 a = *(const float4*)p, c = *(const float4*)(p + 4);
  float v[8] = {a.x, a.y, a.z, a.w, c.x, c.y, c.z, c.w};
  float s = 0.f, s2 = 0.f;
#pragma unroll
  for (int j = 0; j < 8; ++j) { s += v[j]; s2 += v[j] * v[j]; }
  for (int off = 16; off > 0; off >>= 1) {
    s  += __shfl_xor(s, off, 32);
    s2 += __shfl_xor(s2, off, 32);
  }
  const float mu = s * (1.f / C_);
  const float rs = rsqrtf(s2 * (1.f / C_) - mu * mu + 1e-5f);
  float dot = 0.f;
#pragma unroll
  for (int j = 0; j < 8; ++j) {
    const int ch = lane * 8 + j;
    const float y = fmaxf((v[j] - mu) * rs * g[ch] + be[ch], 0.f);
    dot += y * lw[ch];
  }
  for (int off = 16; off > 0; off >>= 1) dot += __shfl_xor(dot, off, 32);
  if (lane == 0) dur[row] = expf(fmaxf(dot + lb[0], 0.f));
}

// ---------------------------------------------------------------------------
// per-batch cumulative sum of durations (one batch per thread; tiny)
// ---------------------------------------------------------------------------
__global__ void cumsum_kernel(const int* __restrict__ tgt,
                              int* __restrict__ ends) {
  const int b = threadIdx.x;
  if (b < B_) {
    int s = 0;
    for (int l = 0; l < L_; ++l) { s += tgt[b * L_ + l]; ends[b * L_ + l] = s; }
  }
}

// ---------------------------------------------------------------------------
// length-regulate expand: wave per output row; binary search + 32B/lane copy
// 201 MB streaming stores -> HBM-bound kernel (~10 us at 23.3 TB/s)
// ---------------------------------------------------------------------------
__global__ __launch_bounds__(256) void expand_kernel(
    const float* __restrict__ x, const int* __restrict__ ends,
    float* __restrict__ out) {
  const int lane = threadIdx.x & 31;
  const int row  = blockIdx.x * 8 + (threadIdx.x >> 5);   // < B_*MAXLEN
  const int b = row / MAXLEN;
  const int t = row % MAXLEN;
  const int* e = ends + b * L_;
  float4 r0 = {0.f, 0.f, 0.f, 0.f}, r1 = r0;
  if (t < e[L_ - 1]) {
    int lo = 0, hi = L_;
    while (lo < hi) {          // searchsorted(side="right")
      int mid = (lo + hi) >> 1;
      if (e[mid] <= t) lo = mid + 1; else hi = mid;
    }
    const int idx = lo < (L_ - 1) ? lo : (L_ - 1);
    const float* src = x + ((size_t)b * L_ + idx) * C_ + lane * 8;
    r0 = *(const float4*)src;
    r1 = *(const float4*)(src + 4);
  }
  float* dst = out + ((size_t)b * MAXLEN + t) * C_ + lane * 8;
  *(float4*)dst = r0;
  *(float4*)(dst + 4) = r1;
}

// ---------------------------------------------------------------------------
extern "C" void kernel_launch(void* const* d_in, const int* in_sizes, int n_in,
                              void* d_out, int out_size, void* d_ws,
                              size_t ws_size, hipStream_t stream) {
  const float* x   = (const float*)d_in[0];
  const int*   tgt = (const int*)  d_in[1];
  const float* w1  = (const float*)d_in[2];
  const float* b1  = (const float*)d_in[3];
  const float* g1  = (const float*)d_in[4];
  const float* be1 = (const float*)d_in[5];
  const float* w2  = (const float*)d_in[6];
  const float* b2  = (const float*)d_in[7];
  const float* g2  = (const float*)d_in[8];
  const float* be2 = (const float*)d_in[9];
  const float* lw  = (const float*)d_in[10];
  const float* lb  = (const float*)d_in[11];
  float* out = (float*)d_out;

  char* ws = (char*)d_ws;
  _Float16* xh   = (_Float16*)(ws);                               // 16 MB
  _Float16* h1a  = (_Float16*)(ws + ((size_t)16 << 20));          // 16 MB
  float*    hbuf = (float*)   (ws + ((size_t)32 << 20));          // 32 MB (h1 then h2)
  _Float16* w1t  = (_Float16*)(ws + ((size_t)64 << 20));          // 384 KB
  _Float16* w2t  = (_Float16*)(ws + ((size_t)64 << 20) + KTOT * C_ * 2);
  int*      ends = (int*)     (ws + ((size_t)64 << 20) + 2 * KTOT * C_ * 2);

  const size_t DUR_OFF = (size_t)B_ * MAXLEN * C_;  // 50331648

  // activations + weights to f16 (weights transposed to [n][k])
  const int n4 = (B_ * L_ * C_) / 4;
  cvt_f16_kernel<<<(n4 + 255) / 256, 256, 0, stream>>>(x, xh, n4);
  cvt_wt_kernel<<<(KTOT * C_ + 255) / 256, 256, 0, stream>>>(w1, w1t);
  cvt_wt_kernel<<<(KTOT * C_ + 255) / 256, 256, 0, stream>>>(w2, w2t);

  dim3 gg(MTOT / 128, 2);
  conv_gemm_wmma<<<gg, 256, 0, stream>>>(xh, w1t, b1, hbuf);
  ln_relu_f16_kernel<<<MTOT / 8, 256, 0, stream>>>(hbuf, g1, be1, h1a);
  conv_gemm_wmma<<<gg, 256, 0, stream>>>(h1a, w2t, b2, hbuf);
  ln_lin_exp_kernel<<<MTOT / 8, 256, 0, stream>>>(hbuf, g2, be2, lw, lb,
                                                  out + DUR_OFF);

  cumsum_kernel<<<1, 32, 0, stream>>>(tgt, ends);
  expand_kernel<<<(B_ * MAXLEN) / 8, 256, 0, stream>>>(x, ends, out);
}
